// MLCABlock_62577673503622
// MI455X (gfx1250) — matrix operand order repaired and checked by reading, not compiled, not run on hardware
//
#include <hip/hip_runtime.h>
#include <hip/hip_bf16.h>
#include <stdint.h>

#define DMODEL 512
#define NNODES 64
#define NHEADS 8
#define DFF    2048
#define DKH    64
#define BATCH  8
#define SEQLEN 128
#define NTOK   (BATCH * SEQLEN * NNODES) /* 65536 tokens */

typedef __attribute__((ext_vector_type(16))) __bf16 v16bf;
typedef __attribute__((ext_vector_type(8)))  float  v8f;

// ---------------- bf16 helpers (RNE) ----------------
__device__ __forceinline__ unsigned short f2bf(float f) {
  unsigned u = __float_as_uint(f);
  u += 0x7FFFu + ((u >> 16) & 1u);
  return (unsigned short)(u >> 16);
}
__device__ __forceinline__ float bf2f(unsigned short h) {
  return __uint_as_float(((unsigned)h) << 16);
}

// ---------------- elementwise f32 -> bf16 ----------------
__global__ __launch_bounds__(256) void cvt_f32_bf16(const float* __restrict__ in,
                                                    unsigned short* __restrict__ out, int n) {
  int i = blockIdx.x * 256 + threadIdx.x;
  if (i < n) out[i] = f2bf(in[i]);
}

// ---------------- bias concat (qkv) ----------------
__global__ void bias_cat3(const float* a, const float* b, const float* c, float* o) {
  int i = blockIdx.x * 256 + threadIdx.x;
  if (i >= 1536) return;
  o[i] = (i < 512) ? a[i] : ((i < 1024) ? b[i - 512] : c[i - 1024]);
}

// ---------------- weight pre-swizzle into WMMA B-fragment order ----------------
// dword[ ((nt*Kt + kt)*32 + lane)*8 + v ] = { W[nt*16+(lane&15)][k], W[nt*16+(lane&15)][k+1] },
//   k = kt*32 + 8*(lane>>4) + 2*(v&3) + 16*(v>>2)
// so a wave's B fragment for tile (nt,kt) is 8 contiguous dwords per lane.
__global__ __launch_bounds__(256) void pack_w(const float* __restrict__ W,
                                              uint32_t* __restrict__ out, int Kt, int ntOff) {
  int nt   = blockIdx.x / Kt;
  int kt   = blockIdx.x % Kt;
  int v    = threadIdx.x & 7;
  int lane = threadIdx.x >> 3;
  int K    = Kt * 32;
  int row0 = nt * 16 + (lane & 15);
  int k    = kt * 32 + (lane >> 4) * 8 + (v & 3) * 2 + (v >> 2) * 16;
  unsigned lo = f2bf(W[(size_t)row0 * K + k]);
  unsigned hh = f2bf(W[(size_t)row0 * K + k + 1]);
  out[((((size_t)(nt + ntOff) * Kt) + kt) * 32 + lane) * 8 + v] = lo | (hh << 16);
}

// ---------------- WMMA bf16 GEMM: C[M,N] = A[M,K] * W[N,K]^T + bias ----------------
union FragAB { v16bf v; uint4 q[2]; };
union FragC  { v8f v; float f[8]; };

__device__ __forceinline__ void load_a(FragAB& a, const unsigned short* __restrict__ arow,
                                       int kt, int hi) {
  const uint4* ap = (const uint4*)(arow + kt * 32);
  a.q[0] = ap[hi];       // lanes 0-15: K 0..7   | lanes 16-31: K 8..15
  a.q[1] = ap[hi + 2];   // lanes 0-15: K 16..23 | lanes 16-31: K 24..31
}

__device__ __forceinline__ void load_b4(FragAB* b, const uint32_t* __restrict__ Wp,
                                        int Kt, int kt, int lane, int ng) {
#pragma unroll
  for (int j = 0; j < 4; ++j) {
    const uint4* bp =
        (const uint4*)(Wp + ((((size_t)(ng * 4 + j) * Kt + kt) * 32 + lane) << 3));
    b[j].q[0] = bp[0];
    b[j].q[1] = bp[1];
  }
}

__device__ __forceinline__ void mma4(FragC* c, const FragAB& a, const FragAB* b) {
#pragma unroll
  for (int j = 0; j < 4; ++j)
    c[j].v = __builtin_amdgcn_wmma_f32_16x16x32_bf16(false, a.v, false, b[j].v, (short)0,
                                                     c[j].v, false, false);
}

// Block = 128 rows x 64 cols: 8 waves, one 16-row tile each, ALL sharing the same
// B tiles (identical per-lane B addresses across waves -> L0 broadcast).
// K loop is software-pipelined with two register buffers so loads for kt+1 are in
// flight while kt's four WMMAs execute.
__global__ __launch_bounds__(256) void gemm_bf16_wmma(
    const unsigned short* __restrict__ A, const uint32_t* __restrict__ Wp,
    const float* __restrict__ bias, float* __restrict__ outF,
    unsigned short* __restrict__ outH, int M, int N, int K, int ostride, int relu) {
  int wv   = threadIdx.x >> 5;
  int lane = threadIdx.x & 31;
  int nGroups = N >> 6;                       // 64 output cols per block
  int bm = blockIdx.x / nGroups;              // 128-row block index
  int ng = blockIdx.x % nGroups;
  int mt = bm * 8 + wv;                       // this wave's 16-row tile
  if (mt * 16 >= M) return;
  int ml = lane & 15, hi = lane >> 4;
  const unsigned short* arow = A + (size_t)(mt * 16 + ml) * K;
  int Kt = K >> 5;

  FragC c[4];
#pragma unroll
  for (int j = 0; j < 4; ++j)
#pragma unroll
    for (int r = 0; r < 8; ++r) c[j].f[r] = 0.f;

  FragAB a0, a1, b0[4], b1[4];
  load_a(a0, arow, 0, hi);
  load_b4(b0, Wp, Kt, 0, lane, ng);

  for (int kt = 0; kt < Kt; kt += 2) {        // Kt is even (16 or 64)
    __builtin_prefetch(arow + (kt + 2) * 32, 0, 0);  // global_prefetch_b8, A stream
    load_a(a1, arow, kt + 1, hi);
    load_b4(b1, Wp, Kt, kt + 1, lane, ng);
    mma4(c, a0, b0);                          // overlaps with buf1 loads
    if (kt + 2 < Kt) {
      load_a(a0, arow, kt + 2, hi);
      load_b4(b0, Wp, Kt, kt + 2, lane, ng);
    }
    mma4(c, a1, b1);                          // overlaps with buf0 loads
  }

#pragma unroll
  for (int j = 0; j < 4; ++j) {
    int col = ng * 64 + j * 16 + ml;
    float bcol = bias ? bias[col] : 0.f;
#pragma unroll
    for (int r = 0; r < 8; ++r) {
      int m = mt * 16 + r + 8 * hi;           // C/D layout: vgpr r, lane -> (m, n)
      float val = c[j].f[r] + bcol;
      if (relu) val = fmaxf(val, 0.f);
      if (outH) outH[(size_t)m * ostride + col] = f2bf(val);
      else      outF[(size_t)m * ostride + col] = val;
    }
  }
}

// ---------------- attention core (online softmax, K/V in LDS) ----------------
template <int SEQ, bool CAUSAL>
__device__ __forceinline__ void attn_core(const unsigned short* __restrict__ qkv,
                                          unsigned short* __restrict__ ctx, int base,
                                          int step, int h, unsigned short* ks,
                                          unsigned short* vs) {
  const int tid = threadIdx.x;  // one row (query position) per thread
  const float scale = 0.125f;   // 1/sqrt(64)
  {
    size_t r = (size_t)(base + step * tid) * 1536;
    const uint4* kr = (const uint4*)(qkv + r + 512 + h * 64);
    const uint4* vr = (const uint4*)(qkv + r + 1024 + h * 64);
    uint4* kd = (uint4*)(ks + tid * 64);
    uint4* vd = (uint4*)(vs + tid * 64);
#pragma unroll
    for (int cpy = 0; cpy < 8; ++cpy) { kd[cpy] = kr[cpy]; vd[cpy] = vr[cpy]; }
  }
  __syncthreads();
  float qf[DKH];
  {
    const unsigned short* qr = qkv + (size_t)(base + step * tid) * 1536 + h * 64;
#pragma unroll
    for (int j = 0; j < DKH; ++j) qf[j] = bf2f(qr[j]) * scale;
  }
  float m = -1e30f, l = 0.f;
  float acc[DKH];
#pragma unroll
  for (int j = 0; j < DKH; ++j) acc[j] = 0.f;
  int tmax = CAUSAL ? tid : (SEQ - 1);
  for (int t = 0; t <= tmax; ++t) {
    const unsigned short* kt = ks + t * 64;
    float d = 0.f;
#pragma unroll
    for (int j = 0; j < DKH; ++j) d += qf[j] * bf2f(kt[j]);
    float nm   = fmaxf(m, d);
    float eold = __expf(m - nm);
    float ecur = __expf(d - nm);
    l = l * eold + ecur;
    const unsigned short* vt = vs + t * 64;
#pragma unroll
    for (int j = 0; j < DKH; ++j) acc[j] = acc[j] * eold + ecur * bf2f(vt[j]);
    m = nm;
  }
  float inv = 1.f / l;
  unsigned short* orow = ctx + (size_t)(base + step * tid) * 512 + h * 64;
#pragma unroll
  for (int j = 0; j < DKH; ++j) orow[j] = f2bf(acc[j] * inv);
}

__global__ __launch_bounds__(128) void attn_temporal(const unsigned short* __restrict__ qkv,
                                                     unsigned short* __restrict__ ctx) {
  __shared__ unsigned short ks[SEQLEN * 64], vs[SEQLEN * 64];
  int blk = blockIdx.x;                 // B*N*H = 4096
  int h = blk & 7, n = (blk >> 3) & 63, b = blk >> 9;
  attn_core<SEQLEN, true>(qkv, ctx, b * SEQLEN * NNODES + n, NNODES, h, ks, vs);
}

__global__ __launch_bounds__(64) void attn_spatial(const unsigned short* __restrict__ qkv,
                                                   unsigned short* __restrict__ ctx) {
  __shared__ unsigned short ks[NNODES * 64], vs[NNODES * 64];
  int blk = blockIdx.x;                 // B*S*H = 8192
  int h = blk & 7, s = (blk >> 3) & 127, b = blk >> 10;
  attn_core<NNODES, false>(qkv, ctx, (b * SEQLEN + s) * NNODES, 1, h, ks, vs);
}

// ---------------- layer-norm helpers / fused residual kernels ----------------
template <int NT>
__device__ __forceinline__ void ln_block(float* row, float* red, const float* g,
                                         const float* bb, int tid) {
  __syncthreads();
  float s = 0.f;
  for (int j = tid; j < 512; j += NT) s += row[j];
  red[tid] = s; __syncthreads();
  for (int o = NT / 2; o > 0; o >>= 1) { if (tid < o) red[tid] += red[tid + o]; __syncthreads(); }
  float mu = red[0] * (1.f / 512.f);
  __syncthreads();
  float s2 = 0.f;
  for (int j = tid; j < 512; j += NT) { float d = row[j] - mu; s2 += d * d; }
  red[tid] = s2; __syncthreads();
  for (int o = NT / 2; o > 0; o >>= 1) { if (tid < o) red[tid] += red[tid + o]; __syncthreads(); }
  float rstd = rsqrtf(red[0] * (1.f / 512.f) + 1e-5f);
  __syncthreads();
  for (int j = tid; j < 512; j += NT) row[j] = (row[j] - mu) * rstd * g[j] + bb[j];
  __syncthreads();
}

// x1 = LN(x + temp_out); also emit bf16(x1 + node_emb) for the in_proj GEMM
__global__ __launch_bounds__(256) void k1_resln(const float* __restrict__ x,
                                                const float* __restrict__ t,
                                                const float* __restrict__ g,
                                                const float* __restrict__ bb,
                                                const float* __restrict__ nemb,
                                                float* __restrict__ x1,
                                                unsigned short* __restrict__ xe) {
  __shared__ float row[512]; __shared__ float red[256];
  size_t o = (size_t)blockIdx.x * 512; int tid = threadIdx.x;
  for (int j = tid; j < 512; j += 256) row[j] = x[o + j] + t[o + j];
  ln_block<256>(row, red, g, bb, tid);
  int n = blockIdx.x & (NNODES - 1);
  for (int j = tid; j < 512; j += 256) {
    float y = row[j];
    x1[o + j] = y;
    xe[o + j] = f2bf(y + nemb[(size_t)n * 512 + j]);
  }
}

// spatial_out = LN(x1+sattn); x2 = LN(x1+spatial_out); write x2 (f32 + bf16)
__global__ __launch_bounds__(256) void k2_spatial_merge(
    const float* __restrict__ x1, const float* __restrict__ sattn,
    const float* __restrict__ g1, const float* __restrict__ b1,
    const float* __restrict__ g2, const float* __restrict__ b2,
    float* __restrict__ x2out, unsigned short* __restrict__ xbf) {
  __shared__ float row[512]; __shared__ float xs[512]; __shared__ float red[256];
  size_t o = (size_t)blockIdx.x * 512; int tid = threadIdx.x;
  for (int j = tid; j < 512; j += 256) {
    float a = x1[o + j];
    xs[j] = a;
    row[j] = a + sattn[o + j];
  }
  ln_block<256>(row, red, g1, b1, tid);       // row = spatial_out
  for (int j = tid; j < 512; j += 256) row[j] = xs[j] + row[j];
  ln_block<256>(row, red, g2, b2, tid);       // row = x2
  for (int j = tid; j < 512; j += 256) {
    x2out[o + j] = row[j];
    xbf[o + j]  = f2bf(row[j]);
  }
}

// out = LN(x2 + ffn)
__global__ __launch_bounds__(256) void k3_finln(const float* __restrict__ x2,
                                                const float* __restrict__ h,
                                                const float* __restrict__ g,
                                                const float* __restrict__ bb,
                                                float* __restrict__ out) {
  __shared__ float row[512]; __shared__ float red[256];
  size_t o = (size_t)blockIdx.x * 512; int tid = threadIdx.x;
  for (int j = tid; j < 512; j += 256) row[j] = x2[o + j] + h[o + j];
  ln_block<256>(row, red, g, bb, tid);
  for (int j = tid; j < 512; j += 256) out[o + j] = row[j];
}

// ---------------- host-side orchestration ----------------
extern "C" void kernel_launch(void* const* d_in, const int* in_sizes, int n_in,
                              void* d_out, int out_size, void* d_ws, size_t ws_size,
                              hipStream_t stream) {
  (void)in_sizes; (void)n_in; (void)out_size; (void)ws_size;
  const float* x          = (const float*)d_in[0];
  const float* Wq         = (const float*)d_in[1];
  const float* bq         = (const float*)d_in[2];
  const float* Wk         = (const float*)d_in[3];
  const float* bk         = (const float*)d_in[4];
  const float* Wv         = (const float*)d_in[5];
  const float* bv         = (const float*)d_in[6];
  const float* Wo         = (const float*)d_in[7];
  const float* bo         = (const float*)d_in[8];
  const float* ln1_g      = (const float*)d_in[9];
  const float* ln1_b      = (const float*)d_in[10];
  const float* node_emb   = (const float*)d_in[11];
  const float* in_proj_w  = (const float*)d_in[12];
  const float* in_proj_b  = (const float*)d_in[13];
  const float* out_proj_w = (const float*)d_in[14];
  const float* out_proj_b = (const float*)d_in[15];
  const float* lnsp_g     = (const float*)d_in[16];
  const float* lnsp_b     = (const float*)d_in[17];
  const float* ln2_g      = (const float*)d_in[18];
  const float* ln2_b      = (const float*)d_in[19];
  const float* fc1_w      = (const float*)d_in[20];
  const float* fc1_b      = (const float*)d_in[21];
  const float* fc2_w      = (const float*)d_in[22];
  const float* fc2_b      = (const float*)d_in[23];
  const float* lnffn_g    = (const float*)d_in[24];
  const float* lnffn_b    = (const float*)d_in[25];
  float* out = (float*)d_out;

  char* ws = (char*)d_ws;
  size_t off = 0;
  auto take = [&](size_t bytes) {
    void* p = ws + off;
    off = (off + bytes + 255) & ~(size_t)255;
    return p;
  };
  unsigned short* in_bf  = (unsigned short*)take((size_t)NTOK * 512 * 2);   //  64 MB
  unsigned short* act_bf = (unsigned short*)take((size_t)NTOK * 2048 * 2);  // 256 MB
  float*          gout   = (float*)take((size_t)NTOK * 512 * 4);            // 128 MB
  float*          r1     = (float*)take((size_t)NTOK * 512 * 4);            // 128 MB
  uint32_t*       packW  = (uint32_t*)take((size_t)4 << 20);                //   4 MB
  float*          bias3  = (float*)take(1536 * 4);

  auto gemm = [&](const unsigned short* A, const uint32_t* Wp, const float* bias,
                  float* oF, unsigned short* oH, int M, int N, int K, int ostride,
                  int relu) {
    int blocks = (M / 128) * (N / 64);  // 8 waves per block, one 16-row tile each
    gemm_bf16_wmma<<<blocks, 256, 0, stream>>>(A, Wp, bias, oF, oH, M, N, K, ostride, relu);
  };

  // ---- stage 1: temporal causal attention ----
  cvt_f32_bf16<<<(NTOK * 512) / 256, 256, 0, stream>>>(x, in_bf, NTOK * 512);
  pack_w<<<32 * 16, 256, 0, stream>>>(Wq, packW, 16, 0);
  pack_w<<<32 * 16, 256, 0, stream>>>(Wk, packW, 16, 32);
  pack_w<<<32 * 16, 256, 0, stream>>>(Wv, packW, 16, 64);
  bias_cat3<<<6, 256, 0, stream>>>(bq, bk, bv, bias3);
  gemm(in_bf, packW, bias3, nullptr, act_bf, NTOK, 1536, 512, 1536, 0);       // qkv
  attn_temporal<<<BATCH * NNODES * NHEADS, SEQLEN, 0, stream>>>(act_bf, in_bf);
  pack_w<<<32 * 16, 256, 0, stream>>>(Wo, packW, 16, 0);
  gemm(in_bf, packW, bo, gout, nullptr, NTOK, 512, 512, 512, 0);              // temp_out
  k1_resln<<<NTOK, 256, 0, stream>>>(x, gout, ln1_g, ln1_b, node_emb, r1, in_bf);

  // ---- stage 2: spatial attention ----
  pack_w<<<96 * 16, 256, 0, stream>>>(in_proj_w, packW, 16, 0);
  gemm(in_bf, packW, in_proj_b, nullptr, act_bf, NTOK, 1536, 512, 1536, 0);   // qkv
  attn_spatial<<<BATCH * SEQLEN * NHEADS, NNODES, 0, stream>>>(act_bf, in_bf);
  pack_w<<<32 * 16, 256, 0, stream>>>(out_proj_w, packW, 16, 0);
  gemm(in_bf, packW, out_proj_b, gout, nullptr, NTOK, 512, 512, 512, 0);      // sattn
  k2_spatial_merge<<<NTOK, 256, 0, stream>>>(r1, gout, lnsp_g, lnsp_b, ln2_g, ln2_b,
                                             r1, in_bf);

  // ---- stage 3: FFN ----
  pack_w<<<128 * 16, 256, 0, stream>>>(fc1_w, packW, 16, 0);
  gemm(in_bf, packW, fc1_b, nullptr, act_bf, NTOK, DFF, 512, DFF, 1);         // relu(fc1)
  pack_w<<<32 * 64, 256, 0, stream>>>(fc2_w, packW, 64, 0);
  gemm(act_bf, packW, fc2_b, gout, nullptr, NTOK, 512, DFF, 512, 0);          // fc2
  k3_finln<<<NTOK, 256, 0, stream>>>(r1, gout, lnffn_g, lnffn_b, out);
}
